// TimeMixing_14602888806421
// MI455X (gfx1250) — matrix operand order repaired and checked by reading, not compiled, not run on hardware
//
#include <hip/hip_runtime.h>
#include <hip/hip_bf16.h>
#include <math.h>

typedef __bf16 bf16_t;
typedef __attribute__((ext_vector_type(16))) __bf16 v16bf;
typedef __attribute__((ext_vector_type(8)))  __bf16 v8bf;
typedef __attribute__((ext_vector_type(8)))  float  v8f;
typedef __attribute__((ext_vector_type(4)))  int    v4i;

#define B_      2
#define T_      2048
#define C_      2048
#define HEAD_   64
#define H_      (C_ / HEAD_)
#define BT_     (B_ * T_)
#define DMIX_   32
#define DDECAY_ 64
#define EPS_    6.4e-4f   // 1e-5 * 8^2

#if defined(__HIP_DEVICE_COMPILE__) && \
    __has_builtin(__builtin_amdgcn_global_load_async_to_lds_b128) && \
    __has_builtin(__builtin_amdgcn_s_wait_asynccnt)
#define STAGE_ASYNC 1
typedef __attribute__((address_space(1))) v4i* gv4i_p;
typedef __attribute__((address_space(3))) v4i* lv4i_p;
#else
#define STAGE_ASYNC 0
#endif

// ---------------------------------------------------------------------------
// K0: convert fp32 weight (K x N row-major) -> bf16 in WMMA-B fragment-tiled
// layout. Tile = 32(K) x 16(N), stored as 512 contiguous bf16 in fragment
// order: f = lane*16 + j, lane L -> n = L%16, k = (L/16)*8 + j (j<8), 16 + (L/16)*8 + (j-8) (j>=8).
// tileIndex = ktile * (N/16) + ntile  (consecutive ntiles contiguous).
// ---------------------------------------------------------------------------
__global__ void tile_weight_bf16(const float* __restrict__ W, bf16_t* __restrict__ Wt) {
    const int ntiles = C_ / 16;
    const size_t total = (size_t)C_ * C_;
    for (size_t o = (size_t)blockIdx.x * blockDim.x + threadIdx.x; o < total;
         o += (size_t)gridDim.x * blockDim.x) {
        int    f     = (int)(o & 511);
        size_t tile  = o >> 9;
        int    ntile = (int)(tile % ntiles);
        int    ktile = (int)(tile / ntiles);
        int    lane  = f >> 4;
        int    j     = f & 15;
        int    k     = ktile * 32 + (j < 8 ? (lane >> 4) * 8 + j : 16 + (lane >> 4) * 8 + (j - 8));
        int    n     = ntile * 16 + (lane & 15);
        Wt[o] = (bf16_t)W[(size_t)k * C_ + n];
    }
}

// ---------------------------------------------------------------------------
// K1: token shift: xx = x[t-1]-x[t] (zero pad), xxx = x + xx*time_maa_x
// ---------------------------------------------------------------------------
__global__ void shift_mix(const float* __restrict__ x, const float* __restrict__ maa_x,
                          float* __restrict__ xx, float* __restrict__ xxx) {
    size_t idx = (size_t)blockIdx.x * blockDim.x + threadIdx.x;
    if (idx >= (size_t)BT_ * C_) return;
    int   c    = (int)(idx % C_);
    int   t    = (int)((idx / C_) % T_);
    float xv   = x[idx];
    float prev = (t > 0) ? x[idx - C_] : 0.0f;
    float d    = prev - xv;
    xx[idx]  = d;
    xxx[idx] = xv + d * maa_x[c];
}

// ---------------------------------------------------------------------------
// K2: mraw = tanh(xxx @ time_maa_w1)   (BT x 160, K = C)
// ---------------------------------------------------------------------------
__global__ void mix_w1(const float* __restrict__ xxx, const float* __restrict__ w1,
                       float* __restrict__ mraw) {
    __shared__ float row[C_];
    const int bt = blockIdx.x;
    for (int k = threadIdx.x; k < C_; k += blockDim.x) row[k] = xxx[(size_t)bt * C_ + k];
    __syncthreads();
    const int col = threadIdx.x;
    if (col < 5 * DMIX_) {
        float acc = 0.f;
        for (int k = 0; k < C_; ++k) acc += row[k] * w1[(size_t)k * (5 * DMIX_) + col];
        mraw[(size_t)bt * (5 * DMIX_) + col] = tanhf(acc);
    }
}

// ---------------------------------------------------------------------------
// K3: per-branch mixes; xw fp32 (feeds decay MLP), xk/xv/xr/xg -> bf16.
// ---------------------------------------------------------------------------
__global__ void branch_mix(const float* __restrict__ x,   const float* __restrict__ xx,
                           const float* __restrict__ mraw, const float* __restrict__ w2,
                           const float* __restrict__ maa_w, const float* __restrict__ maa_k,
                           const float* __restrict__ maa_v, const float* __restrict__ maa_r,
                           const float* __restrict__ maa_g,
                           float* __restrict__ xw,
                           bf16_t* __restrict__ xk, bf16_t* __restrict__ xv,
                           bf16_t* __restrict__ xr, bf16_t* __restrict__ xg) {
    __shared__ float m[5 * DMIX_];
    const int bt = blockIdx.x;
    if (threadIdx.x < 5 * DMIX_) m[threadIdx.x] = mraw[(size_t)bt * (5 * DMIX_) + threadIdx.x];
    __syncthreads();
    const size_t base = (size_t)bt * C_;
    for (int c = threadIdx.x; c < C_; c += blockDim.x) {
        float acc0 = 0.f, acc1 = 0.f, acc2 = 0.f, acc3 = 0.f, acc4 = 0.f;
#pragma unroll 8
        for (int d = 0; d < DMIX_; ++d) {
            acc0 += m[0 * DMIX_ + d] * w2[((size_t)(0 * DMIX_ + d)) * C_ + c];
            acc1 += m[1 * DMIX_ + d] * w2[((size_t)(1 * DMIX_ + d)) * C_ + c];
            acc2 += m[2 * DMIX_ + d] * w2[((size_t)(2 * DMIX_ + d)) * C_ + c];
            acc3 += m[3 * DMIX_ + d] * w2[((size_t)(3 * DMIX_ + d)) * C_ + c];
            acc4 += m[4 * DMIX_ + d] * w2[((size_t)(4 * DMIX_ + d)) * C_ + c];
        }
        float xval = x[base + c], xd = xx[base + c];
        xw[base + c] = xval + xd * (maa_w[c] + acc0);
        xk[base + c] = (bf16_t)(xval + xd * (maa_k[c] + acc1));
        xv[base + c] = (bf16_t)(xval + xd * (maa_v[c] + acc2));
        xr[base + c] = (bf16_t)(xval + xd * (maa_r[c] + acc3));
        xg[base + c] = (bf16_t)(xval + xd * (maa_g[c] + acc4));
    }
}

// ---------------------------------------------------------------------------
// K4: ddt = tanh(xw @ time_decay_w1)   (BT x 64)
// ---------------------------------------------------------------------------
__global__ void decay_w1(const float* __restrict__ xw, const float* __restrict__ dw1,
                         float* __restrict__ ddt) {
    __shared__ float row[C_];
    const int bt = blockIdx.x;
    for (int k = threadIdx.x; k < C_; k += blockDim.x) row[k] = xw[(size_t)bt * C_ + k];
    __syncthreads();
    if (threadIdx.x < DDECAY_) {
        float acc = 0.f;
        for (int k = 0; k < C_; ++k) acc += row[k] * dw1[(size_t)k * DDECAY_ + threadIdx.x];
        ddt[(size_t)bt * DDECAY_ + threadIdx.x] = tanhf(acc);
    }
}

// ---------------------------------------------------------------------------
// K5: w = time_decay + ddt @ time_decay_w2;  dec = exp(-exp(w))
// ---------------------------------------------------------------------------
__global__ void decay_out(const float* __restrict__ ddt, const float* __restrict__ dw2,
                          const float* __restrict__ tdecay, float* __restrict__ dec) {
    __shared__ float dd[DDECAY_];
    const int bt = blockIdx.x;
    if (threadIdx.x < DDECAY_) dd[threadIdx.x] = ddt[(size_t)bt * DDECAY_ + threadIdx.x];
    __syncthreads();
    for (int c = threadIdx.x; c < C_; c += blockDim.x) {
        float acc = tdecay[c];
#pragma unroll 16
        for (int d = 0; d < DDECAY_; ++d) acc += dd[d] * dw2[(size_t)d * C_ + c];
        dec[(size_t)bt * C_ + c] = __expf(-__expf(acc));
    }
}

// ---------------------------------------------------------------------------
// K6: bf16 WMMA GEMM: Cout(M,N) = A(M,K) @ Btiled(K,N), f32 accumulate.
// Block 256 thr = 8 waves; block tile 128x128; wave tile 32x64 (2x4 WMMA).
// A tiles double-buffered in LDS, staged with async global->LDS copies
// (ASYNCcnt) when available; B read from fragment-pre-tiled weights.
// ---------------------------------------------------------------------------
#define GTM 128
#define GTN 128
#define GTK 32

__global__ __launch_bounds__(256)
void gemm_bf16_wmma(const bf16_t* __restrict__ A, const bf16_t* __restrict__ Bt,
                    float* __restrict__ Cout, int M, int N, int K) {
    __shared__ __align__(16) bf16_t As[2][GTM][GTK + 8];   // 80B row stride, 2 buffers
    const int tid    = threadIdx.x;
    const int lane   = tid & 31;
    const int wave   = tid >> 5;
    const int mw     = wave & 3;                 // 0..3 -> M quadrant (32 rows)
    const int nw     = wave >> 2;                // 0..1 -> N half (64 cols)
    const int blockM = blockIdx.y * GTM;
    const int blockN = blockIdx.x * GTN;
    const int ntiles = N >> 4;
    const int fm     = lane & 15;                // A fragment row
    const int fk     = (lane >> 4) * 8;          // A fragment K half (halves)

    v8f acc[2][4];
#pragma unroll
    for (int i = 0; i < 2; ++i)
#pragma unroll
        for (int j = 0; j < 4; ++j) acc[i][j] = v8f{};

    auto stage = [&](int kt, int buf) {
#pragma unroll
        for (int it = 0; it < 2; ++it) {
            int i  = tid + it * 256;             // 0..511 chunk id (16B each)
            int r  = i >> 2;
            int cc = (i & 3) * 8;
            const bf16_t* g = A + (size_t)(blockM + r) * K + kt * GTK + cc;
            bf16_t*       l = &As[buf][r][cc];
#if STAGE_ASYNC
            __builtin_amdgcn_global_load_async_to_lds_b128((gv4i_p)g, (lv4i_p)l, 0, 0);
#else
            *(v8bf*)l = *(const v8bf*)g;
            __builtin_prefetch(g + GTK, 0, 1);
#endif
        }
    };

    const int nk = K / GTK;
    stage(0, 0);
#if STAGE_ASYNC
    __builtin_amdgcn_s_wait_asynccnt(0);
#endif
    __syncthreads();

    for (int kt = 0; kt < nk; ++kt) {
        const int buf = kt & 1;
        if (kt + 1 < nk) stage(kt + 1, buf ^ 1);   // overlap next-tile staging

        // A fragments (2 m-tiles) from LDS
        v16bf a0, a1;
        {
            int  r0 = mw * 32 + fm;
            v8bf lo = *(const v8bf*)(&As[buf][r0][fk]);
            v8bf hi = *(const v8bf*)(&As[buf][r0][16 + fk]);
            a0 = __builtin_shufflevector(lo, hi, 0,1,2,3,4,5,6,7,8,9,10,11,12,13,14,15);
            int r1 = r0 + 16;
            lo = *(const v8bf*)(&As[buf][r1][fk]);
            hi = *(const v8bf*)(&As[buf][r1][16 + fk]);
            a1 = __builtin_shufflevector(lo, hi, 0,1,2,3,4,5,6,7,8,9,10,11,12,13,14,15);
        }
        // B fragments (4 n-tiles) from pre-tiled weights (coalesced 32B/lane)
        v16bf bfrag[4];
        {
            int nt0 = (blockN + nw * 64) >> 4;
            const bf16_t* bbase = Bt + (((size_t)kt * ntiles + nt0) << 9) + lane * 16;
#pragma unroll
            for (int j = 0; j < 4; ++j) {
                v8bf lo = *(const v8bf*)(bbase + j * 512);
                v8bf hi = *(const v8bf*)(bbase + j * 512 + 8);
                bfrag[j] = __builtin_shufflevector(lo, hi, 0,1,2,3,4,5,6,7,8,9,10,11,12,13,14,15);
            }
        }
#pragma unroll
        for (int j = 0; j < 4; ++j) {
            acc[0][j] = __builtin_amdgcn_wmma_f32_16x16x32_bf16(false, a0, false, bfrag[j],
                                                               (short)0, acc[0][j], false, false);
            acc[1][j] = __builtin_amdgcn_wmma_f32_16x16x32_bf16(false, a1, false, bfrag[j],
                                                               (short)0, acc[1][j], false, false);
        }
#if STAGE_ASYNC
        __builtin_amdgcn_s_wait_asynccnt(0);
#endif
        __syncthreads();
    }

    // C/D layout: VGPR r -> M = r (lanes 0-15) or r+8 (lanes 16-31); N = lane%16
    const int mrow = blockM + mw * 32 + (lane >> 4) * 8;
    const int ncol = blockN + nw * 64 + (lane & 15);
#pragma unroll
    for (int i = 0; i < 2; ++i)
#pragma unroll
        for (int j = 0; j < 4; ++j)
#pragma unroll
            for (int r = 0; r < 8; ++r)
                Cout[(size_t)(mrow + i * 16 + r) * N + ncol + j * 16] = acc[i][j][r];
}

// ---------------------------------------------------------------------------
// K7: WKV6 recurrence. One block per (b,h); thread i holds state column
// S[j][i] (j = 0..63) in registers; decay = exp(-exp(w)) precomputed.
// ---------------------------------------------------------------------------
__global__ __launch_bounds__(HEAD_)
void wkv6_kernel(const float* __restrict__ r, const float* __restrict__ k,
                 const float* __restrict__ v, const float* __restrict__ dec,
                 const float* __restrict__ u, float* __restrict__ y) {
    const int b = blockIdx.x / H_;
    const int h = blockIdx.x % H_;
    const int i = threadIdx.x;                   // 0..63
    __shared__ float rs[HEAD_], ks[HEAD_], ws[HEAD_], us[HEAD_];
    us[i] = u[h * HEAD_ + i];
    float S[HEAD_];
#pragma unroll
    for (int j = 0; j < HEAD_; ++j) S[j] = 0.f;
    const size_t base = (size_t)b * T_ * C_ + (size_t)h * HEAD_ + i;
    for (int t = 0; t < T_; ++t) {
        const size_t idx = base + (size_t)t * C_;
        __syncthreads();
        rs[i] = r[idx];
        ks[i] = k[idx];
        ws[i] = dec[idx];
        const float vi = v[idx];
        __syncthreads();
        float yi = 0.f;
#pragma unroll
        for (int j = 0; j < HEAD_; ++j) {
            float a = ks[j] * vi;
            yi += rs[j] * (S[j] + us[j] * a);
            S[j] = S[j] * ws[j] + a;
        }
        y[idx] = yi;
    }
}

// ---------------------------------------------------------------------------
// K8: GroupNorm over heads + SiLU(g) multiply, pack bf16 for final GEMM.
// ---------------------------------------------------------------------------
__global__ __launch_bounds__(HEAD_)
void gnorm_silu(const float* __restrict__ y, const float* __restrict__ graw,
                const float* __restrict__ lnw, const float* __restrict__ lnb,
                bf16_t* __restrict__ a2) {
    const int bt = blockIdx.x / H_;
    const int h  = blockIdx.x % H_;
    const int i  = threadIdx.x;
    __shared__ float s1[HEAD_], s2[HEAD_];
    const size_t idx = (size_t)bt * C_ + (size_t)h * HEAD_ + i;
    const float val = y[idx];
    s1[i] = val;
    s2[i] = val * val;
    __syncthreads();
    for (int off = HEAD_ / 2; off > 0; off >>= 1) {
        if (i < off) { s1[i] += s1[i + off]; s2[i] += s2[i + off]; }
        __syncthreads();
    }
    const float mu  = s1[0] * (1.f / HEAD_);
    const float var = s2[0] * (1.f / HEAD_) - mu * mu;
    const float yn  = (val - mu) * rsqrtf(var + EPS_) * lnw[h * HEAD_ + i] + lnb[h * HEAD_ + i];
    const float g   = graw[idx];
    const float sg  = g / (1.f + __expf(-g));
    a2[idx] = (bf16_t)(yn * sg);
}

// ---------------------------------------------------------------------------
// Launch
// ---------------------------------------------------------------------------
extern "C" void kernel_launch(void* const* d_in, const int* in_sizes, int n_in,
                              void* d_out, int out_size, void* d_ws, size_t ws_size,
                              hipStream_t stream) {
    const float* x      = (const float*)d_in[0];
    const float* maa_x  = (const float*)d_in[1];
    const float* maa_w  = (const float*)d_in[2];
    const float* maa_k  = (const float*)d_in[3];
    const float* maa_v  = (const float*)d_in[4];
    const float* maa_r  = (const float*)d_in[5];
    const float* maa_g  = (const float*)d_in[6];
    const float* w1     = (const float*)d_in[7];
    const float* w2     = (const float*)d_in[8];
    const float* tdecay = (const float*)d_in[9];
    const float* dw1    = (const float*)d_in[10];
    const float* dw2    = (const float*)d_in[11];
    const float* faaaa  = (const float*)d_in[12];
    const float* W_r    = (const float*)d_in[13];
    const float* W_k    = (const float*)d_in[14];
    const float* W_v    = (const float*)d_in[15];
    const float* W_g    = (const float*)d_in[16];
    const float* W_o    = (const float*)d_in[17];
    const float* ln_w   = (const float*)d_in[18];
    const float* ln_b   = (const float*)d_in[19];
    float* out = (float*)d_out;

    // bump allocator on d_ws
    char* p = (char*)d_ws;
    auto alloc = [&](size_t bytes) -> void* {
        void* r = (void*)p;
        p += (bytes + 255) & ~(size_t)255;
        return r;
    };
    const size_t BTC = (size_t)BT_ * C_;
    bf16_t* wr_t  = (bf16_t*)alloc((size_t)C_ * C_ * sizeof(bf16_t));
    bf16_t* wk_t  = (bf16_t*)alloc((size_t)C_ * C_ * sizeof(bf16_t));
    bf16_t* wv_t  = (bf16_t*)alloc((size_t)C_ * C_ * sizeof(bf16_t));
    bf16_t* wg_t  = (bf16_t*)alloc((size_t)C_ * C_ * sizeof(bf16_t));
    bf16_t* wo_t  = (bf16_t*)alloc((size_t)C_ * C_ * sizeof(bf16_t));
    float*  xx    = (float*)alloc(BTC * sizeof(float));
    float*  xxx   = (float*)alloc(BTC * sizeof(float));   // reused as y after K6
    float*  mraw  = (float*)alloc((size_t)BT_ * 5 * DMIX_ * sizeof(float));
    float*  xw    = (float*)alloc(BTC * sizeof(float));   // reused as graw after K4
    bf16_t* xk    = (bf16_t*)alloc(BTC * sizeof(bf16_t));
    bf16_t* xv    = (bf16_t*)alloc(BTC * sizeof(bf16_t));
    bf16_t* xr    = (bf16_t*)alloc(BTC * sizeof(bf16_t));
    bf16_t* xg    = (bf16_t*)alloc(BTC * sizeof(bf16_t));
    float*  ddt   = (float*)alloc((size_t)BT_ * DDECAY_ * sizeof(float));
    float*  dec   = (float*)alloc(BTC * sizeof(float));
    float*  rbuf  = (float*)alloc(BTC * sizeof(float));
    float*  kbuf  = (float*)alloc(BTC * sizeof(float));
    float*  vbuf  = (float*)alloc(BTC * sizeof(float));
    bf16_t* a2    = (bf16_t*)alloc(BTC * sizeof(bf16_t));
    float*  ybuf  = xxx;   // alias: xxx no longer needed after K2
    float*  graw  = xw;    // alias: xw consumed by K4 before g-GEMM writes

    // K0: weight tiling (fp32 -> bf16 fragment-tiled)
    tile_weight_bf16<<<4096, 256, 0, stream>>>(W_r, wr_t);
    tile_weight_bf16<<<4096, 256, 0, stream>>>(W_k, wk_t);
    tile_weight_bf16<<<4096, 256, 0, stream>>>(W_v, wv_t);
    tile_weight_bf16<<<4096, 256, 0, stream>>>(W_g, wg_t);
    tile_weight_bf16<<<4096, 256, 0, stream>>>(W_o, wo_t);

    // K1: token shift
    shift_mix<<<(unsigned)((BTC + 255) / 256), 256, 0, stream>>>(x, maa_x, xx, xxx);
    // K2: mixing MLP stage 1
    mix_w1<<<BT_, 256, 0, stream>>>(xxx, w1, mraw);
    // K3: per-branch mixes
    branch_mix<<<BT_, 256, 0, stream>>>(x, xx, mraw, w2, maa_w, maa_k, maa_v, maa_r,
                                        maa_g, xw, xk, xv, xr, xg);
    // K4/K5: decay path
    decay_w1<<<BT_, 256, 0, stream>>>(xw, dw1, ddt);
    decay_out<<<BT_, 256, 0, stream>>>(ddt, dw2, tdecay, dec);

    // K6: big projections (WMMA)
    dim3 ggrid(C_ / GTN, BT_ / GTM);
    gemm_bf16_wmma<<<ggrid, 256, 0, stream>>>(xr, wr_t, rbuf, BT_, C_, C_);
    gemm_bf16_wmma<<<ggrid, 256, 0, stream>>>(xk, wk_t, kbuf, BT_, C_, C_);
    gemm_bf16_wmma<<<ggrid, 256, 0, stream>>>(xv, wv_t, vbuf, BT_, C_, C_);
    gemm_bf16_wmma<<<ggrid, 256, 0, stream>>>(xg, wg_t, graw, BT_, C_, C_);

    // K7: WKV6 recurrence
    wkv6_kernel<<<B_ * H_, HEAD_, 0, stream>>>(rbuf, kbuf, vbuf, dec, faaaa, ybuf);

    // K8: groupnorm + silu(g) -> bf16
    gnorm_silu<<<BT_ * H_, HEAD_, 0, stream>>>(ybuf, graw, ln_w, ln_b, a2);

    // K9: output projection (WMMA) -> d_out (fp32)
    gemm_bf16_wmma<<<ggrid, 256, 0, stream>>>(a2, wo_t, out, BT_, C_, C_);
}
